// PlabelingZOGIN_41351945126322
// MI455X (gfx1250) — matrix operand order repaired and checked by reading, not compiled, not run on hardware
//
#include <hip/hip_runtime.h>
#include <hip/hip_bf16.h>

// ---------------------------------------------------------------------------
// Types for CDNA5 WMMA (wave32): V_WMMA_F32_16X16X32_BF16
// ---------------------------------------------------------------------------
typedef __attribute__((ext_vector_type(16))) __bf16 v16bf;
typedef __attribute__((ext_vector_type(8)))  float  v8f;

union FragU {
    uint4 u[2];
    v16bf v;
};

static __device__ __forceinline__ __bf16 f2bf(float f) {
    union { float f; unsigned u; } a;
    a.f = f;
    unsigned r = (a.u + 0x7FFFu + ((a.u >> 16) & 1u)) >> 16;  // RNE
    unsigned short s = (unsigned short)r;
    return __builtin_bit_cast(__bf16, s);
}

// A-matrix fragment (16x32 bf16, M x K), row-major source with 'stride' elems.
// Lane L: m = L&15, half = L>>4. elems 0..7 = K[k0+half*8 ..], 8..15 = K[k0+16+half*8 ..]
static __device__ __forceinline__ v16bf load_a_frag(const __bf16* __restrict__ tile,
                                                    int stride, int lane, int k0) {
    int m = lane & 15;
    int half = lane >> 4;
    const __bf16* p = tile + m * stride + k0;
    FragU f;
    f.u[0] = *(const uint4*)(p + half * 8);
    f.u[1] = *(const uint4*)(p + 16 + half * 8);
    return f.v;
}

// B-matrix fragment (32x16 bf16, K x N) from transposed weights Wt[out][in] (row len K).
// Lane L: n = ntile*16 + (L&15), half = L>>4. elems 0..7 = K[k0+half*8 ..], 8..15 = K[k0+16+half*8 ..]
static __device__ __forceinline__ v16bf load_b_frag(const __bf16* __restrict__ wt,
                                                    int K, int lane, int ntile, int k0) {
    int n = (ntile << 4) + (lane & 15);
    const __bf16* p = wt + (long)n * K + k0;
    FragU f;
    f.u[0] = *(const uint4*)(p + ((lane >> 4) << 3));
    f.u[1] = *(const uint4*)(p + 16 + ((lane >> 4) << 3));
    return f.v;
}

static __device__ __forceinline__ v8f wmma_bf16(v16bf a, v16bf b, v8f c) {
    return __builtin_amdgcn_wmma_f32_16x16x32_bf16(false, a, false, b, (short)0, c,
                                                   false, false);
}

// K=128 GEMM panel: preload ALL fragments (4 A from LDS, 4 B from global), then
// issue 4 back-to-back chained WMMAs (D->C accumulate: no hazard NOPs needed).
// This lets the hardware clause the loads and hide their latency once, instead
// of a load->wait->wmma lockstep per K-step.
static __device__ __forceinline__ v8f gemm_k128(const __bf16* __restrict__ Atile, int astride,
                                                const __bf16* __restrict__ Wt, int K,
                                                int lane, int wave, int kbase, v8f acc) {
    v16bf a0 = load_a_frag(Atile, astride, lane, kbase + 0);
    v16bf a1 = load_a_frag(Atile, astride, lane, kbase + 32);
    v16bf a2 = load_a_frag(Atile, astride, lane, kbase + 64);
    v16bf a3 = load_a_frag(Atile, astride, lane, kbase + 96);
    v16bf b0 = load_b_frag(Wt, K, lane, wave, kbase + 0);
    v16bf b1 = load_b_frag(Wt, K, lane, wave, kbase + 32);
    v16bf b2 = load_b_frag(Wt, K, lane, wave, kbase + 64);
    v16bf b3 = load_b_frag(Wt, K, lane, wave, kbase + 96);
    acc = wmma_bf16(a0, b0, acc);
    acc = wmma_bf16(a1, b1, acc);
    acc = wmma_bf16(a2, b2, acc);
    acc = wmma_bf16(a3, b3, acc);
    return acc;
}

// ---------------------------------------------------------------------------
// Weight prep: fp32 [K][Nout] -> bf16 transposed [Nout][K]
// ---------------------------------------------------------------------------
__global__ void transpose_bf16_kernel(const float* __restrict__ src,
                                      __bf16* __restrict__ dst, int K, int Nout) {
    int t = blockIdx.x * 256 + threadIdx.x;
    if (t >= K * Nout) return;
    int k = t / Nout, n = t % Nout;
    dst[n * K + k] = f2bf(src[t]);
}

// ---------------------------------------------------------------------------
// Embedding: x[c][n][:] = z_table[(z[n]==c+1)?1:0][:]
// ---------------------------------------------------------------------------
__global__ void embed_kernel(const int* __restrict__ z,
                             const float* __restrict__ z_table,
                             float* __restrict__ x, int N) {
    long t = (long)blockIdx.x * 256 + threadIdx.x;
    if (t >= (long)2 * N * 32) return;
    int h4 = (int)(t & 31) * 4;
    long cn = t >> 5;
    int n = (int)(cn % N);
    int c = (int)(cn / N);
    int row = (z[n] == c + 1) ? 1 : 0;
    *(float4*)(x + cn * 128 + h4) = *(const float4*)(z_table + row * 128 + h4);
}

// ---------------------------------------------------------------------------
// Edge scatter: agg[c][dst][h] += x[c][src][h]   (f32 global atomics)
// ---------------------------------------------------------------------------
__global__ void scatter_kernel(const float* __restrict__ x, float* __restrict__ agg,
                               const int* __restrict__ src, const int* __restrict__ dst,
                               int E, int N) {
    long t = (long)blockIdx.x * 256 + threadIdx.x;
    if (t >= (long)2 * E * 32) return;
    int h4 = (int)(t & 31) * 4;
    long ce = t >> 5;
    int e = (int)(ce % E);
    int c = (int)(ce / E);
    int s = src[e], d = dst[e];
    float4 v = *(const float4*)(x + ((long)c * N + s) * 128 + h4);
    float* p = agg + ((long)c * N + d) * 128 + h4;
    atomicAdd(p + 0, v.x);
    atomicAdd(p + 1, v.y);
    atomicAdd(p + 2, v.z);
    atomicAdd(p + 3, v.w);
}

// ---------------------------------------------------------------------------
// Fused GIN layer: h = relu(relu((x+agg)@W1+b1)@W2+b2); x = LN(h)*g+b
// One block = 16 rows of the combined [2N,128] activation matrix.
// 256 threads = 8 waves; wave w owns output columns [16w,16w+16).
// ---------------------------------------------------------------------------
__global__ __launch_bounds__(256) void mlp_layer_kernel(
    float* __restrict__ x, const float* __restrict__ agg,
    const __bf16* __restrict__ W1t, const float* __restrict__ b1,
    const __bf16* __restrict__ W2t, const float* __restrict__ b2,
    const float* __restrict__ lng, const float* __restrict__ lnb,
    __bf16* __restrict__ xsjk, int l) {
    __shared__ __align__(16) __bf16 Xs[16 * 136];
    __shared__ __align__(16) __bf16 Hs[16 * 136];
    __shared__ __align__(16) float  Fs[16 * 132];

    int tid = threadIdx.x;
    long r0 = (long)blockIdx.x * 16;

    // Phase A: stage bf16(x + agg) tile in LDS
    for (int i = tid; i < 16 * 128; i += 256) {
        int rr = i >> 7, cc = i & 127;
        long gi = (r0 + rr) * 128 + cc;
        Xs[rr * 136 + cc] = f2bf(x[gi] + agg[gi]);
    }
    __syncthreads();

    int wave = tid >> 5, lane = tid & 31;
    int colb = wave * 16 + (lane & 15);
    int rbase = (lane >> 4) * 8;

    // GEMM1: [16x128] x [128x16(per wave)]
    v8f acc = {};
    acc = gemm_k128(Xs, 136, W1t, 128, lane, wave, 0, acc);
    float bias1 = b1[l * 128 + colb];
    for (int v = 0; v < 8; ++v) {
        float h = acc[v] + bias1;
        Hs[(rbase + v) * 136 + colb] = f2bf(h > 0.f ? h : 0.f);
    }
    __syncthreads();

    // GEMM2
    v8f acc2 = {};
    acc2 = gemm_k128(Hs, 136, W2t, 128, lane, wave, 0, acc2);
    float bias2 = b2[l * 128 + colb];
    for (int v = 0; v < 8; ++v) {
        float h = acc2[v] + bias2;
        Fs[(rbase + v) * 132 + colb] = h > 0.f ? h : 0.f;
    }
    __syncthreads();

    // LayerNorm: 16 threads per row (one half-wave group), shfl_xor reduce.
    int row = tid >> 4;
    int grp = tid & 15;
    float s = 0.f, ss = 0.f;
    for (int i = 0; i < 8; ++i) {
        float v = Fs[row * 132 + grp * 8 + i];
        s += v;
        ss += v * v;
    }
    for (int m = 8; m >= 1; m >>= 1) {
        s += __shfl_xor(s, m, 16);
        ss += __shfl_xor(ss, m, 16);
    }
    float mean = s * (1.f / 128.f);
    float var = ss * (1.f / 128.f) - mean * mean;
    float rinv = rsqrtf(var + 1e-5f);
    long rowg = r0 + row;
    for (int i = 0; i < 8; ++i) {
        int c = grp * 8 + i;
        float v = (Fs[row * 132 + c] - mean) * rinv * lng[l * 128 + c] + lnb[l * 128 + c];
        x[rowg * 128 + c] = v;                         // in-place: block owns its rows
        xsjk[rowg * 384 + l * 128 + c] = f2bf(v);      // JK concat buffer (bf16)
    }
}

// ---------------------------------------------------------------------------
// lin0: nodeout[n] = sum_c relu(xsjk[c][n][0:384] @ lin0_W + b0)
// One block = 16 nodes, K = 384 (3 K-panels of 128), loops both channels.
// ---------------------------------------------------------------------------
__global__ __launch_bounds__(256) void lin0_kernel(
    const __bf16* __restrict__ xsjk, const __bf16* __restrict__ W0t,
    const float* __restrict__ b0, float* __restrict__ nodeout, int N) {
    __shared__ __align__(16) __bf16 Xs[16 * 392];
    int tid = threadIdx.x;
    int wave = tid >> 5, lane = tid & 31;
    long n0 = (long)blockIdx.x * 16;
    int colb = wave * 16 + (lane & 15);
    int rbase = (lane >> 4) * 8;
    float bias = b0[colb];

    float res[8];
    for (int v = 0; v < 8; ++v) res[v] = 0.f;

    for (int c = 0; c < 2; ++c) {
        __syncthreads();
        for (int i = tid; i < 16 * 384; i += 256) {
            int rr = i / 384, cc = i % 384;
            Xs[rr * 392 + cc] = xsjk[((long)c * N + n0 + rr) * 384 + cc];
        }
        __syncthreads();
        v8f acc = {};
        acc = gemm_k128(Xs, 392, W0t, 384, lane, wave, 0, acc);
        acc = gemm_k128(Xs, 392, W0t, 384, lane, wave, 128, acc);
        acc = gemm_k128(Xs, 392, W0t, 384, lane, wave, 256, acc);
        for (int v = 0; v < 8; ++v) {
            float h = acc[v] + bias;
            res[v] += h > 0.f ? h : 0.f;
        }
    }
    for (int v = 0; v < 8; ++v)
        nodeout[(n0 + rbase + v) * 128 + colb] = res[v];
}

// ---------------------------------------------------------------------------
// Segment-sum pooling + counts (f32 atomics into [G,128] / [G])
// ---------------------------------------------------------------------------
__global__ void pool_kernel(const float* __restrict__ nodeout,
                            const int* __restrict__ batch,
                            float* __restrict__ pooled, float* __restrict__ cnt, int N) {
    long t = (long)blockIdx.x * 256 + threadIdx.x;
    if (t >= (long)N * 32) return;
    int h4 = (int)(t & 31) * 4;
    int n = (int)(t >> 5);
    int g = batch[n];
    float4 v = *(const float4*)(nodeout + (long)n * 128 + h4);
    float* p = pooled + (long)g * 128 + h4;
    atomicAdd(p + 0, v.x);
    atomicAdd(p + 1, v.y);
    atomicAdd(p + 2, v.z);
    atomicAdd(p + 3, v.w);
    if ((t & 31) == 0) atomicAdd(&cnt[g], 1.0f);
}

// ---------------------------------------------------------------------------
// Head: y = relu((pooled/max(cnt,1)) @ linA + bA)   (G=128 rows, WMMA)
// ---------------------------------------------------------------------------
__global__ __launch_bounds__(256) void head_kernel(
    const float* __restrict__ pooled, const float* __restrict__ cnt,
    const __bf16* __restrict__ WAt, const float* __restrict__ bA,
    float* __restrict__ y) {
    __shared__ __align__(16) __bf16 Xs[16 * 136];
    int tid = threadIdx.x;
    int g0 = blockIdx.x * 16;
    for (int i = tid; i < 16 * 128; i += 256) {
        int rr = i >> 7, cc = i & 127;
        float c = cnt[g0 + rr];
        c = c > 1.f ? c : 1.f;
        Xs[rr * 136 + cc] = f2bf(pooled[(g0 + rr) * 128 + cc] / c);
    }
    __syncthreads();
    int wave = tid >> 5, lane = tid & 31;
    int colb = wave * 16 + (lane & 15);
    int rbase = (lane >> 4) * 8;
    v8f acc = {};
    acc = gemm_k128(Xs, 136, WAt, 128, lane, wave, 0, acc);
    float bias = bA[colb];
    for (int v = 0; v < 8; ++v) {
        float h = acc[v] + bias;
        y[(g0 + rbase + v) * 128 + colb] = h > 0.f ? h : 0.f;
    }
}

// ---------------------------------------------------------------------------
// Final: out[g] = y[g] . linB_W + bB
// ---------------------------------------------------------------------------
__global__ void final_kernel(const float* __restrict__ y, const float* __restrict__ wB,
                             const float* __restrict__ bB, float* __restrict__ out) {
    int g = blockIdx.x * 64 + threadIdx.x;
    if (g >= 128) return;
    float s = 0.f;
    for (int h = 0; h < 128; ++h) s += y[g * 128 + h] * wB[h];
    out[g] = s + bB[0];
}

// ---------------------------------------------------------------------------
// Launch
// ---------------------------------------------------------------------------
extern "C" void kernel_launch(void* const* d_in, const int* in_sizes, int n_in,
                              void* d_out, int out_size, void* d_ws, size_t ws_size,
                              hipStream_t stream) {
    const int N = 50000, E = 800000, H = 128, L = 3, G = 128;

    const int*   z       = (const int*)d_in[0];
    const int*   edge    = (const int*)d_in[1];   // [2][E]
    const int*   batch   = (const int*)d_in[2];
    const float* z_table = (const float*)d_in[3];
    const float* W1s     = (const float*)d_in[4];
    const float* b1s     = (const float*)d_in[5];
    const float* W2s     = (const float*)d_in[6];
    const float* b2s     = (const float*)d_in[7];
    const float* lng     = (const float*)d_in[8];
    const float* lnb     = (const float*)d_in[9];
    const float* lin0_W  = (const float*)d_in[10];
    const float* lin0_b  = (const float*)d_in[11];
    const float* linA_W  = (const float*)d_in[12];
    const float* linA_b  = (const float*)d_in[13];
    const float* linB_W  = (const float*)d_in[14];
    const float* linB_b  = (const float*)d_in[15];
    float* out = (float*)d_out;

    // ---- workspace carve ----
    char* p = (char*)d_ws;
    auto alloc = [&](size_t bytes) -> char* {
        char* r = p;
        p += (bytes + 255) & ~(size_t)255;
        return r;
    };
    float*  x       = (float*)alloc((size_t)2 * N * H * 4);       // 51.2 MB
    float*  agg     = (float*)alloc((size_t)2 * N * H * 4);       // 51.2 MB
    __bf16* xsjk    = (__bf16*)alloc((size_t)2 * N * L * H * 2);  // 76.8 MB
    float*  nodeout = (float*)alloc((size_t)N * H * 4);           // 25.6 MB
    float*  pooled  = (float*)alloc((size_t)G * H * 4);
    float*  cnt     = (float*)alloc((size_t)G * 4);
    float*  y       = (float*)alloc((size_t)G * H * 4);
    __bf16* W1t     = (__bf16*)alloc((size_t)L * H * H * 2);
    __bf16* W2t     = (__bf16*)alloc((size_t)L * H * H * 2);
    __bf16* W0t     = (__bf16*)alloc((size_t)L * H * H * 2);      // [128][384]
    __bf16* WAt     = (__bf16*)alloc((size_t)H * H * 2);

    // ---- weight prep (bf16 transpose) ----
    {
        int nb = (H * H + 255) / 256;
        for (int l = 0; l < L; ++l) {
            transpose_bf16_kernel<<<nb, 256, 0, stream>>>(W1s + l * H * H, W1t + l * H * H, H, H);
            transpose_bf16_kernel<<<nb, 256, 0, stream>>>(W2s + l * H * H, W2t + l * H * H, H, H);
        }
        transpose_bf16_kernel<<<(L * H * H + 255) / 256, 256, 0, stream>>>(lin0_W, W0t, L * H, H);
        transpose_bf16_kernel<<<nb, 256, 0, stream>>>(linA_W, WAt, H, H);
    }

    // ---- embedding ----
    {
        long t = (long)2 * N * 32;
        embed_kernel<<<(int)((t + 255) / 256), 256, 0, stream>>>(z, z_table, x, N);
    }

    // ---- GIN layers ----
    const int* src = edge;
    const int* dst = edge + E;
    for (int l = 0; l < L; ++l) {
        hipMemsetAsync(agg, 0, (size_t)2 * N * H * 4, stream);
        long st = (long)2 * E * 32;
        scatter_kernel<<<(int)((st + 255) / 256), 256, 0, stream>>>(x, agg, src, dst, E, N);
        mlp_layer_kernel<<<2 * N / 16, 256, 0, stream>>>(
            x, agg, W1t + l * H * H, b1s, W2t + l * H * H, b2s, lng, lnb, xsjk, l);
    }

    // ---- lin0 + channel sum ----
    lin0_kernel<<<N / 16, 256, 0, stream>>>(xsjk, W0t, lin0_b, nodeout, N);

    // ---- mean pool ----
    hipMemsetAsync(pooled, 0, (size_t)G * H * 4, stream);
    hipMemsetAsync(cnt, 0, (size_t)G * 4, stream);
    {
        long t = (long)N * 32;
        pool_kernel<<<(int)((t + 255) / 256), 256, 0, stream>>>(nodeout, batch, pooled, cnt, N);
    }

    // ---- head GEMM + final projection ----
    head_kernel<<<G / 16, 256, 0, stream>>>(pooled, cnt, WAt, linA_b, y);
    final_kernel<<<2, 64, 0, stream>>>(y, linB_W, linB_b, out);
}
